// GATv2Encoder_39109972198186
// MI455X (gfx1250) — matrix operand order repaired and checked by reading, not compiled, not run on hardware
//
#include <hip/hip_runtime.h>
#include <hip/hip_bf16.h>

#define NN    40000
#define EEg   480000
#define EAg   (EEg + NN)
#define IND   64
#define HIDD  128
#define NLAY  4
#define NHEAD 4
#define CCH   32
#define EDIM  4
#define NG    256

#define CDIV(a,b) (((a)+(b)-1)/(b))

typedef __attribute__((ext_vector_type(16))) _Float16 v16h;
typedef __attribute__((ext_vector_type(8)))  _Float16 v8h;
typedef __attribute__((ext_vector_type(8)))  float    v8f;

// ---------- monotonic float <-> uint encoding for atomicMax-based segment max ----------
__device__ __forceinline__ unsigned encf(float x) {
  unsigned u = __float_as_uint(x);
  return (u & 0x80000000u) ? ~u : (u | 0x80000000u);
}
__device__ __forceinline__ float decf(unsigned e) {
  unsigned u = (e & 0x80000000u) ? (e & 0x7FFFFFFFu) : ~e;
  return __uint_as_float(u);
}
// encf(-inf) == 0x007FFFFF
#define ENC_NEG_INF 0x007FFFFFu

// ---------- utility kernels ----------
__global__ void k_fill_u32(unsigned* __restrict__ p, unsigned v, int n) {
  int i = blockIdx.x * blockDim.x + threadIdx.x;
  if (i < n) p[i] = v;
}

__global__ void k_cvt_f16(_Float16* __restrict__ dst, const float* __restrict__ src, int n) {
  int i = blockIdx.x * blockDim.x + threadIdx.x;
  if (i < n) dst[i] = (_Float16)src[i];
}

// ---------- self-loop attr: degree + mean of incoming edge_attr ----------
__global__ void k_deg_loop(const int* __restrict__ dstI, const float* __restrict__ eattr,
                           float* __restrict__ deg, float* __restrict__ loop) {
  int e = blockIdx.x * blockDim.x + threadIdx.x;
  if (e >= EEg) return;
  int d = dstI[e];
  atomicAdd(&deg[d], 1.0f);
  #pragma unroll
  for (int j = 0; j < EDIM; ++j)
    atomicAdd(&loop[d * EDIM + j], eattr[(size_t)e * EDIM + j]);
}

__global__ void k_loop_div(float* __restrict__ loop, const float* __restrict__ deg) {
  int i = blockIdx.x * blockDim.x + threadIdx.x;
  if (i >= NN * EDIM) return;
  loop[i] /= fmaxf(deg[i >> 2], 1.0f);
}

// ---------- WMMA GEMM: Out[M,Nc] = A[M,K](f32->f16) @ B[K,Nc](f16) + bias, opt tanh ----
// Block = 256 threads (8 waves) computes a 128-row strip x Nc=128.
// B is staged TRANSPOSED into LDS once (sBt[n*K+k]) so each lane's 16
// halfs (two runs of 8 K-consecutive, 16 apart) are two aligned ds_load_b128's. Per K-step, one A
// fragment is built and ALL 8 B fragments are prefetched before the 8
// back-to-back WMMAs (single dscnt drain amortized over the WMMA burst).
__global__ void k_gemm_wmma(const float* __restrict__ A, const _Float16* __restrict__ B,
                            const float* __restrict__ bias, float* __restrict__ Out,
                            int M, int K, int Nc, int act) {
  __shared__ _Float16 sBt[HIDD * HIDD];   // up to 128x128 f16 = 32 KB
  // stage B transposed: sBt[n*K + k] = B[k*Nc + n]
  for (int i = threadIdx.x; i < K * Nc; i += blockDim.x) {
    int k = i >> 7, n = i & (HIDD - 1);   // Nc == 128 always here
    sBt[n * K + k] = B[i];
  }
  __syncthreads();

  int wave = threadIdx.x >> 5;            // 0..7
  int lane = threadIdx.x & 31;
  int l15  = lane & 15;
  int hi   = lane >> 4;
  int row0 = blockIdx.x * 128 + wave * 16;
  if (row0 >= M) return;                  // uniform per wave -> EXEC stays all-ones

  v8f acc[8];
  #pragma unroll
  for (int nt = 0; nt < 8; ++nt) acc[nt] = (v8f){};

  for (int k0 = 0; k0 < K; k0 += 32) {
    // A fragment: 16-bit A 16x32 layout (lane=row&15, hi selects K half of 8)
    v16h a;
    const float* arow = A + (size_t)(row0 + l15) * K + k0;
    #pragma unroll
    for (int v = 0; v < 8; ++v) {
      int kk = ((v >> 2) << 4) + (hi << 3) + ((v & 3) << 1);
      float2 f2 = *(const float2*)(arow + kk);
      a[2 * v]     = (_Float16)f2.x;
      a[2 * v + 1] = (_Float16)f2.y;
    }
    // prefetch all 8 B fragments (16 ds_load_b128) before the WMMA burst
    v16h bfrag[8];
    #pragma unroll
    for (int nt = 0; nt < 8; ++nt) {
      const _Float16* bp_ = sBt + (size_t)(nt * 16 + l15) * K + k0 + (hi << 3);
      v8h blo = *(const v8h*)(bp_);
      v8h bhi = *(const v8h*)(bp_ + 16);
      bfrag[nt] = __builtin_shufflevector(blo, bhi, 0, 1, 2, 3, 4, 5, 6, 7,
                                                    8, 9, 10, 11, 12, 13, 14, 15);
    }
    // 8 column tiles share this A fragment; independent accumulator chains
    #pragma unroll
    for (int nt = 0; nt < 8; ++nt) {
      acc[nt] = __builtin_amdgcn_wmma_f32_16x16x32_f16(false, a, false, bfrag[nt],
                                                       (short)0, acc[nt], false, false);
    }
  }

  #pragma unroll
  for (int nt = 0; nt < 8; ++nt) {
    int col = nt * 16 + l15;
    float bv = bias ? bias[col] : 0.0f;
    #pragma unroll
    for (int r = 0; r < 8; ++r) {
      int row = row0 + r + (hi << 3);     // C layout: VGPR r -> M = r + 8*hi
      float o = acc[nt][r] + bv;
      if (act == 1) o = tanhf(o);
      Out[(size_t)row * Nc + col] = o;
    }
  }
}

// ---------- edge attention: logits + segment max (one thread per (edge, head)) ----------
__global__ void k_edge_logits(const int* __restrict__ srcI, const int* __restrict__ dstI,
                              const float* __restrict__ eattr, const float* __restrict__ loop,
                              const float* __restrict__ xl, const float* __restrict__ xr,
                              const float* __restrict__ We, const float* __restrict__ att,
                              float* __restrict__ logits, unsigned* __restrict__ segmax) {
  __shared__ float sWe[EDIM * HIDD];
  __shared__ float sAtt[HIDD];
  for (int i = threadIdx.x; i < EDIM * HIDD; i += blockDim.x) sWe[i] = We[i];
  for (int i = threadIdx.x; i < HIDD; i += blockDim.x)       sAtt[i] = att[i];
  __syncthreads();

  int tid = blockIdx.x * blockDim.x + threadIdx.x;
  if (tid >= EAg * NHEAD) return;
  int e = tid >> 2, hh = tid & 3;
  int s, d; const float* ea;
  if (e < EEg) { s = srcI[e]; d = dstI[e]; ea = eattr + (size_t)e * EDIM; }
  else         { s = d = e - EEg;          ea = loop  + (size_t)(e - EEg) * EDIM; }
  float e0 = ea[0], e1 = ea[1], e2 = ea[2], e3 = ea[3];

  const float* pl = xl + (size_t)s * HIDD + hh * CCH;
  const float* pr = xr + (size_t)d * HIDD + hh * CCH;
  float logit = 0.0f;
  #pragma unroll
  for (int cc = 0; cc < CCH; ++cc) {
    int col = hh * CCH + cc;
    float m = pl[cc] + pr[cc]
            + e0 * sWe[col] + e1 * sWe[HIDD + col]
            + e2 * sWe[2 * HIDD + col] + e3 * sWe[3 * HIDD + col];
    m = (m > 0.0f) ? m : 0.2f * m;       // leaky_relu(0.2)
    logit += m * sAtt[col];
  }
  logits[tid] = logit;
  atomicMax(&segmax[d * NHEAD + hh], encf(logit));
}

// ---------- segment softmax numerator + denominator ----------
__global__ void k_edge_alpha(const int* __restrict__ dstI, float* __restrict__ logits,
                             const unsigned* __restrict__ segmax, float* __restrict__ segsum) {
  int tid = blockIdx.x * blockDim.x + threadIdx.x;
  if (tid >= EAg * NHEAD) return;
  int e = tid >> 2, hh = tid & 3;
  int d = (e < EEg) ? dstI[e] : (e - EEg);
  float a = __expf(logits[tid] - decf(segmax[d * NHEAD + hh]));
  logits[tid] = a;
  atomicAdd(&segsum[d * NHEAD + hh], a);
}

// ---------- message scatter: acc[dst] += xl[src] * alpha ----------
__global__ void k_edge_scatter(const int* __restrict__ srcI, const int* __restrict__ dstI,
                               const float* __restrict__ logits, const float* __restrict__ segsum,
                               const float* __restrict__ xl, float* __restrict__ acc) {
  int tid = blockIdx.x * blockDim.x + threadIdx.x;
  if (tid >= EAg * NHEAD) return;
  int e = tid >> 2, hh = tid & 3;
  int s, d;
  if (e < EEg) { s = srcI[e]; d = dstI[e]; } else { s = d = e - EEg; }
  float alpha = logits[tid] / segsum[d * NHEAD + hh];
  const float* pl = xl + (size_t)s * HIDD + hh * CCH;
  float* pa = acc + (size_t)d * HIDD + hh * CCH;
  #pragma unroll
  for (int cc = 0; cc < CCH; ++cc) atomicAdd(&pa[cc], pl[cc] * alpha);
}

// ---------- per-node: +bo, LayerNorm, SiLU, residual add into h ----------
__global__ void k_post_node(const float* __restrict__ acc, const float* __restrict__ bo,
                            const float* __restrict__ g, const float* __restrict__ b,
                            float* __restrict__ h) {
  __shared__ float red[HIDD];
  int n = blockIdx.x, t = threadIdx.x;
  float v = acc[(size_t)n * HIDD + t] + bo[t];
  red[t] = v; __syncthreads();
  for (int s = HIDD / 2; s > 0; s >>= 1) { if (t < s) red[t] += red[t + s]; __syncthreads(); }
  float mu = red[0] * (1.0f / HIDD);
  __syncthreads();
  float dv = v - mu;
  red[t] = dv * dv; __syncthreads();
  for (int s = HIDD / 2; s > 0; s >>= 1) { if (t < s) red[t] += red[t + s]; __syncthreads(); }
  float var = red[0] * (1.0f / HIDD);
  float y  = dv * rsqrtf(var + 1e-5f) * g[t] + b[t];
  float si = y / (1.0f + __expf(-y));    // silu
  h[(size_t)n * HIDD + t] += si;
}

// ---------- gate MLP second stage + per-graph segment max ----------
__global__ void k_gate_dot(const float* __restrict__ T, const float* __restrict__ Wg2,
                           const float* __restrict__ bg2, const int* __restrict__ batch,
                           float* __restrict__ gate, unsigned* __restrict__ gmax) {
  int n = blockIdx.x * blockDim.x + threadIdx.x;
  if (n >= NN) return;
  const float* r = T + (size_t)n * HIDD;
  float s = bg2[0];
  #pragma unroll 8
  for (int k = 0; k < HIDD; ++k) s += r[k] * Wg2[k];
  gate[n] = s;
  atomicMax(&gmax[batch[n]], encf(s));
}

__global__ void k_gate_exp(const int* __restrict__ batch, float* __restrict__ gate,
                           const unsigned* __restrict__ gmax, float* __restrict__ gsum,
                           float* __restrict__ cnt) {
  int n = blockIdx.x * blockDim.x + threadIdx.x;
  if (n >= NN) return;
  int g = batch[n];
  float a = __expf(gate[n] - decf(gmax[g]));
  gate[n] = a;
  atomicAdd(&gsum[g], a);
  atomicAdd(&cnt[g], 1.0f);
}

__global__ void k_pool(const int* __restrict__ batch, const float* __restrict__ gate,
                       const float* __restrict__ gsum, const float* __restrict__ h,
                       float* __restrict__ pool) {
  int tid = blockIdx.x * blockDim.x + threadIdx.x;
  if (tid >= NN * HIDD) return;
  int n = tid >> 7, c = tid & (HIDD - 1);
  int g = batch[n];
  float w = gate[n] / gsum[g];
  atomicAdd(&pool[(size_t)g * HIDD + c], h[tid] * w);
}

__global__ void k_out(const float* __restrict__ pool, const float* __restrict__ cnt,
                      float* __restrict__ out) {
  int tid = blockIdx.x * blockDim.x + threadIdx.x;
  if (tid >= NG * HIDD) return;
  out[tid] = pool[tid] / fmaxf(cnt[tid >> 7], 1.0f);
}

// =====================================================================
extern "C" void kernel_launch(void* const* d_in, const int* in_sizes, int n_in,
                              void* d_out, int out_size, void* d_ws, size_t ws_size,
                              hipStream_t stream) {
  (void)in_sizes; (void)n_in; (void)out_size; (void)ws_size;

  const float* x     = (const float*)d_in[0];
  const int*   ei    = (const int*)  d_in[1];
  const float* eattr = (const float*)d_in[2];
  const int*   batch = (const int*)  d_in[3];
  const float* Wp    = (const float*)d_in[4];
  const float* bp    = (const float*)d_in[5];
  const float* Wl    = (const float*)d_in[6];
  const float* bl    = (const float*)d_in[7];
  const float* Wr    = (const float*)d_in[8];
  const float* br    = (const float*)d_in[9];
  const float* We    = (const float*)d_in[10];
  const float* att   = (const float*)d_in[11];
  const float* bo    = (const float*)d_in[12];
  const float* ln_g  = (const float*)d_in[13];
  const float* ln_b  = (const float*)d_in[14];
  const float* Wg1   = (const float*)d_in[15];
  const float* bg1   = (const float*)d_in[16];
  const float* Wg2   = (const float*)d_in[17];
  const float* bg2   = (const float*)d_in[18];
  const int* srcI = ei;
  const int* dstI = ei + EEg;

  // ---- workspace carve-up (256B aligned slabs) ----
  char* ws = (char*)d_ws;
  size_t off = 0;
  auto take = [&](size_t bytes) -> char* {
    char* p = ws + off;
    off = (off + bytes + 255) & ~(size_t)255;
    return p;
  };
  float*    h      = (float*)   take((size_t)NN * HIDD * 4);
  float*    xl     = (float*)   take((size_t)NN * HIDD * 4);
  float*    xr     = (float*)   take((size_t)NN * HIDD * 4);
  float*    acc    = (float*)   take((size_t)NN * HIDD * 4);
  float*    logits = (float*)   take((size_t)EAg * NHEAD * 4);
  unsigned* segmax = (unsigned*)take((size_t)NN * NHEAD * 4);
  float*    segsum = (float*)   take((size_t)NN * NHEAD * 4);
  float*    deg    = (float*)   take((size_t)NN * 4);
  float*    loop   = (float*)   take((size_t)NN * EDIM * 4);
  _Float16* wh     = (_Float16*)take((size_t)(8192 + 2 * 4 * 16384 + 16384) * 2);
  float*    gate   = (float*)   take((size_t)NN * 4);
  unsigned* gmax   = (unsigned*)take((size_t)NG * 4);
  float*    gsum   = (float*)   take((size_t)NG * 4);
  float*    cnt    = (float*)   take((size_t)NG * 4);
  float*    pool   = (float*)   take((size_t)NG * HIDD * 4);

  _Float16* WpH  = wh;                   // 64*128
  _Float16* WlH  = WpH + 8192;           // 4 * 128*128
  _Float16* WrH  = WlH + 4 * 16384;      // 4 * 128*128
  _Float16* Wg1H = WrH + 4 * 16384;      // 128*128

  // ---- self-loop attributes (mean of incoming edge_attr) ----
  k_fill_u32<<<CDIV(NN, 256), 256, 0, stream>>>((unsigned*)deg, 0u, NN);
  k_fill_u32<<<CDIV(NN * EDIM, 256), 256, 0, stream>>>((unsigned*)loop, 0u, NN * EDIM);
  k_deg_loop<<<CDIV(EEg, 256), 256, 0, stream>>>(dstI, eattr, deg, loop);
  k_loop_div<<<CDIV(NN * EDIM, 256), 256, 0, stream>>>(loop, deg);

  // ---- cache weights in f16 for WMMA ----
  k_cvt_f16<<<CDIV(8192, 256), 256, 0, stream>>>(WpH, Wp, 8192);
  k_cvt_f16<<<CDIV(4 * 16384, 256), 256, 0, stream>>>(WlH, Wl, 4 * 16384);
  k_cvt_f16<<<CDIV(4 * 16384, 256), 256, 0, stream>>>(WrH, Wr, 4 * 16384);
  k_cvt_f16<<<CDIV(16384, 256), 256, 0, stream>>>(Wg1H, Wg1, 16384);

  const int gblk = CDIV(NN, 128);        // 128-row strip per 256-thread block

  // ---- input projection: h = x @ Wp + bp ----
  k_gemm_wmma<<<gblk, 256, 0, stream>>>(x, WpH, bp, h, NN, IND, HIDD, 0);

  // ---- 4 GATv2 layers ----
  for (int i = 0; i < NLAY; ++i) {
    k_fill_u32<<<CDIV(NN * HIDD, 256), 256, 0, stream>>>((unsigned*)acc, 0u, NN * HIDD);
    k_fill_u32<<<CDIV(NN * NHEAD, 256), 256, 0, stream>>>(segmax, ENC_NEG_INF, NN * NHEAD);
    k_fill_u32<<<CDIV(NN * NHEAD, 256), 256, 0, stream>>>((unsigned*)segsum, 0u, NN * NHEAD);

    k_gemm_wmma<<<gblk, 256, 0, stream>>>(h, WlH + (size_t)i * 16384, bl + i * HIDD,
                                          xl, NN, HIDD, HIDD, 0);
    k_gemm_wmma<<<gblk, 256, 0, stream>>>(h, WrH + (size_t)i * 16384, br + i * HIDD,
                                          xr, NN, HIDD, HIDD, 0);

    int eg = CDIV(EAg * NHEAD, 256);
    k_edge_logits<<<eg, 256, 0, stream>>>(srcI, dstI, eattr, loop, xl, xr,
                                          We + (size_t)i * EDIM * HIDD,
                                          att + (size_t)i * HIDD, logits, segmax);
    k_edge_alpha<<<eg, 256, 0, stream>>>(dstI, logits, segmax, segsum);
    k_edge_scatter<<<eg, 256, 0, stream>>>(srcI, dstI, logits, segsum, xl, acc);

    k_post_node<<<NN, HIDD, 0, stream>>>(acc, bo + i * HIDD, ln_g + i * HIDD,
                                         ln_b + i * HIDD, h);
  }

  // ---- attention pooling ----
  k_gemm_wmma<<<gblk, 256, 0, stream>>>(h, Wg1H, bg1, xl, NN, HIDD, HIDD, 1); // tanh
  k_fill_u32<<<1, 256, 0, stream>>>(gmax, ENC_NEG_INF, NG);
  k_fill_u32<<<1, 256, 0, stream>>>((unsigned*)gsum, 0u, NG);
  k_fill_u32<<<1, 256, 0, stream>>>((unsigned*)cnt, 0u, NG);
  k_fill_u32<<<CDIV(NG * HIDD, 256), 256, 0, stream>>>((unsigned*)pool, 0u, NG * HIDD);

  k_gate_dot<<<CDIV(NN, 256), 256, 0, stream>>>(xl, Wg2, bg2, batch, gate, gmax);
  k_gate_exp<<<CDIV(NN, 256), 256, 0, stream>>>(batch, gate, gmax, gsum, cnt);
  k_pool<<<CDIV(NN * HIDD, 256), 256, 0, stream>>>(batch, gate, gsum, h, pool);
  k_out<<<CDIV(NG * HIDD, 256), 256, 0, stream>>>(pool, cnt, (float*)d_out);
}
